// UnTAN_89172111000351
// MI455X (gfx1250) — matrix operand order repaired and checked by reading, not compiled, not run on hardware
//
#include <hip/hip_runtime.h>
#include <hip/hip_bf16.h>

#define B_  32
#define LQ_ 512
#define LK_ 1024
#define D_  128
#define E_  128
#define H_  8
#define NH_ 512
#define DK_ 16
#define NC_ (LK_ / 32)   // 32 key chunks

typedef __attribute__((ext_vector_type(16))) _Float16 v16h;
typedef __attribute__((ext_vector_type(8)))  _Float16 v8h;
typedef __attribute__((ext_vector_type(8)))  float    v8f;
typedef __attribute__((ext_vector_type(4)))  int      v4i;

// ---- CDNA5 async global->LDS (guarded; sync v8h copy fallback) -------------
#if defined(__has_builtin)
#  if __has_builtin(__builtin_amdgcn_global_load_async_to_lds_b128)
#    define HAVE_ASYNC_LDS 1
#  endif
#endif
#ifndef HAVE_ASYNC_LDS
#  define HAVE_ASYNC_LDS 0
#endif

#if HAVE_ASYNC_LDS
#  define GPTR(p) ((v4i*)(p))
#  define LPTR(p) ((v4i*)(p))
#  if __has_builtin(__builtin_amdgcn_s_wait_asynccnt)
#    define WAIT_ASYNC(n) __builtin_amdgcn_s_wait_asynccnt(n)
#  else
#    define WAIT_ASYNC(n) asm volatile("s_wait_asynccnt %0" ::"i"(n))
#  endif
#endif

// ---------------------------------------------------------------------------
// Kernel A: periodic time embedding + E x E projection + head split -> f16.
// One block per (b, l); thread o computes output channel o.
// pad!=0: rows padded to 32 halfs with zeros in [16..31] (WMMA K-padding
// comes from memory instead of VALU masking).
// ---------------------------------------------------------------------------
__global__ void embed_project_kernel(const float* __restrict__ t_in,
                                     const float* __restrict__ w_t,
                                     const float* __restrict__ b_t,
                                     const float* __restrict__ W,
                                     _Float16* __restrict__ dst,
                                     int L, float scale, int pad) {
  __shared__ float emb[E_];
  const int bl = blockIdx.x;
  const int o  = threadIdx.x;
  const float tv = t_in[bl];
  emb[o] = __sinf(fmaf(tv, w_t[o], b_t[o]));
  __syncthreads();
  const float* wrow = W + (size_t)o * E_;
  float acc = 0.f;
#pragma unroll 8
  for (int e = 0; e < E_; ++e) acc = fmaf(emb[e], wrow[e], acc);
  const int b = bl / L, l = bl % L;
  const int h = o >> 4, dk = o & 15;
  if (pad) {
    const size_t row = ((((size_t)b * H_ + h) * L) + l) * 32;
    dst[row + dk]      = (_Float16)(acc * scale);
    dst[row + 16 + dk] = (_Float16)0.f;
  } else {
    dst[((((size_t)b * H_ + h) * L) + l) * DK_ + dk] = (_Float16)(acc * scale);
  }
}

// ---------------------------------------------------------------------------
// Kernel B: pack mask*value and mask to f16, pre-transposed per 32-key chunk:
// layout [b][chunk][d][kk] so the hot loop stages with straight contiguous
// global->LDS copies (async-friendly) and B-operand reads stay contiguous.
// ---------------------------------------------------------------------------
__global__ void pack_mv_kernel(const float* __restrict__ value,
                               const int* __restrict__ mask,
                               _Float16* __restrict__ mvT,
                               _Float16* __restrict__ mT) {
  size_t i = (size_t)blockIdx.x * blockDim.x + threadIdx.x;
  if (i >= (size_t)B_ * LK_ * D_) return;
  const int d  = (int)(i % D_);
  const size_t t = i / D_;
  const int lk = (int)(t % LK_);
  const int b  = (int)(t / LK_);
  const int c  = lk >> 5, kk = lk & 31;
  const size_t o = ((((size_t)b * NC_ + c) * D_) + d) * 32 + kk;
  const int mk = mask[i];
  mvT[o] = mk ? (_Float16)value[i] : (_Float16)0.f;
  mT[o]  = mk ? (_Float16)1.f      : (_Float16)0.f;
}

__global__ void cvt_w2_kernel(const float* __restrict__ W2,
                              _Float16* __restrict__ w2h) {
  size_t i = (size_t)blockIdx.x * blockDim.x + threadIdx.x;
  if (i < (size_t)NH_ * H_ * D_) w2h[i] = (_Float16)W2[i];
}

// ---------------------------------------------------------------------------
// Kernel C: fused attention. One WG per (b, h, 64-query tile); 8 waves.
// Wave w: 16 q rows (w>>1), 64-channel half (w&1).
// Double-buffered async global->LDS staging of mv/m chunks; per chunk:
// WMMA scores -> exp (row-max cancels in num/den, skipped) -> per-wave LDS
// relayout -> 8 accumulate WMMAs.
// ---------------------------------------------------------------------------
__global__ void __launch_bounds__(256)
attn_kernel(const _Float16* __restrict__ qf16,   // [bh][lq][16]
            const _Float16* __restrict__ kf16,   // [bh][lk][32] zero-padded K
            const _Float16* __restrict__ mvT,    // [b][chunk][d][kk]
            const _Float16* __restrict__ mT,
            _Float16* __restrict__ xf16) {
  __shared__ __align__(64) _Float16 s_mv[2][D_ * 32];   // 2 x 8KB
  __shared__ __align__(64) _Float16 s_m [2][D_ * 32];   // 2 x 8KB
  __shared__ __align__(64) _Float16 s_e [8 * 16 * 32];  // per-wave e tiles 8KB

  const int tid  = threadIdx.x;
  const int lane = tid & 31;
  const int wave = tid >> 5;
  const int lrow = lane & 15;   // M (A/C) or N (B) selector
  const int lhi  = lane >> 4;   // lane half: selects K sub-range

  const int qt = blockIdx.x;
  const int h  = blockIdx.y;
  const int b  = blockIdx.z;
  const int bh = b * H_ + h;

  const int qrow0 = qt * 64 + (wave >> 1) * 16;
  const int d0b   = (wave & 1) * 64;

  // A operand: q tile 16x32 (K 16..31 zero pad in registers).
  v16h aq;
  {
    const _Float16* qp = qf16 + (((size_t)bh * LQ_) + qrow0 + lrow) * DK_ + lhi * 8;
    v8h ql = *(const v8h*)qp;
#pragma unroll
    for (int i = 0; i < 8; ++i) { aq[i] = ql[i]; aq[8 + i] = (_Float16)0.f; }
  }

  const size_t chunk_base = (size_t)b * NC_ * (D_ * 32);
  auto stage = [&](int c, int bb) {
    const _Float16* gmv = mvT + chunk_base + (size_t)c * (D_ * 32);
    const _Float16* gm  = mT  + chunk_base + (size_t)c * (D_ * 32);
#pragma unroll
    for (int r = 0; r < 2; ++r) {
      const int off = (r * 256 + tid) * 8;  // 16B per copy, 16KB total/stage
#if HAVE_ASYNC_LDS
      __builtin_amdgcn_global_load_async_to_lds_b128(GPTR(gmv + off),
                                                     LPTR(&s_mv[bb][off]), 0, 0);
      __builtin_amdgcn_global_load_async_to_lds_b128(GPTR(gm + off),
                                                     LPTR(&s_m[bb][off]), 0, 0);
#else
      *(v8h*)&s_mv[bb][off] = *(const v8h*)(gmv + off);
      *(v8h*)&s_m [bb][off] = *(const v8h*)(gm + off);
#endif
    }
  };

  v8f num[4], den[4];
#pragma unroll
  for (int dt = 0; dt < 4; ++dt) { num[dt] = {}; den[dt] = {}; }

  stage(0, 0);

  for (int c = 0; c < NC_; ++c) {
    const int bb = c & 1;
    if (c + 1 < NC_) stage(c + 1, bb ^ 1);      // overlap next chunk's copies
#if HAVE_ASYNC_LDS
    if (c + 1 < NC_) WAIT_ASYNC(4);             // retire chunk c (4 newest = c+1)
    else             WAIT_ASYNC(0);
#endif
    __syncthreads();                            // chunk c visible to all waves

    if (c + 1 < NC_)                            // global_prefetch_b8 on next k
      __builtin_prefetch(kf16 + (((size_t)bh * LK_) + (c + 1) * 32 + lane) * 32, 0, 1);

    // Scores for 16 q rows x 32 keys; e staged to per-wave LDS as f16.
    // B operand: lane half h holds K=8h..8h+7 (real) and 16+8h..16+8h+7 (zero pad).
#pragma unroll
    for (int t = 0; t < 2; ++t) {
      const _Float16* kp =
          kf16 + (((size_t)bh * LK_) + c * 32 + t * 16 + lrow) * 32 + lhi * 16;
      v8h klo = *(const v8h*)(kp - lhi * 8);
      v8h khi = *(const v8h*)(kp - lhi * 8 + 16);
      v16h bk;
#pragma unroll
      for (int i = 0; i < 8; ++i) { bk[i] = klo[i]; bk[8 + i] = khi[i]; }
      v8f z = {};
      v8f s = __builtin_amdgcn_wmma_f32_16x16x32_f16(false, aq, false, bk,
                                                     (short)0, z, false, false);
      // C layout: VGPR j -> M = j + lhi*8, N = lrow.  e[m][kk] f16.
      _Float16* ep = s_e + wave * 512 + (t * 16 + lrow);
#pragma unroll
      for (int j = 0; j < 8; ++j)
        ep[(j + lhi * 8) * 32] = (_Float16)__expf(s[j]);
    }

    // Re-load e in A layout (same wave wrote it; compiler inserts DS waits).
    v16h ae;
    {
      const _Float16* ep = s_e + wave * 512 + lrow * 32 + lhi * 16;
      v8h lo  = *(const v8h*)ep;
      v8h hi8 = *(const v8h*)(ep + 8);
#pragma unroll
      for (int i = 0; i < 8; ++i) { ae[i] = lo[i]; ae[8 + i] = hi8[i]; }
    }

    // num += e @ mv,  den += e @ m   (4 channel tiles of 16 per wave)
#pragma unroll
    for (int dt = 0; dt < 4; ++dt) {
      const int dcol = d0b + dt * 16 + lrow;
      v16h bmv = *(const v16h*)(&s_mv[bb][0] + (size_t)dcol * 32 + lhi * 16);
      num[dt] = __builtin_amdgcn_wmma_f32_16x16x32_f16(false, ae, false, bmv,
                                                       (short)0, num[dt], false, false);
      v16h bm  = *(const v16h*)(&s_m[bb][0] + (size_t)dcol * 32 + lhi * 16);
      den[dt] = __builtin_amdgcn_wmma_f32_16x16x32_f16(false, ae, false, bm,
                                                       (short)0, den[dt], false, false);
    }
    __syncthreads();  // all waves done with buffer bb before it is re-staged
  }

  // x = num/den -> f16, layout [B, LQ, H*D] (head-major channels).
#pragma unroll
  for (int dt = 0; dt < 4; ++dt) {
#pragma unroll
    for (int j = 0; j < 8; ++j) {
      const int m = j + lhi * 8;
      const size_t o = (((size_t)b * LQ_) + qrow0 + m) * (H_ * D_)
                     + h * D_ + d0b + dt * 16 + lrow;
      xf16[o] = (_Float16)(num[dt][j] / den[dt][j]);
    }
  }
}

// ---------------------------------------------------------------------------
// Kernel D: out = x @ W2^T  ([16384,1024] x [1024,512]), pure WMMA GEMM.
// ---------------------------------------------------------------------------
__global__ void __launch_bounds__(256)
out_gemm_kernel(const _Float16* __restrict__ xf16,
                const _Float16* __restrict__ w2h,
                float* __restrict__ out) {
  const int tid  = threadIdx.x;
  const int lane = tid & 31;
  const int wave = tid >> 5;
  const int lrow = lane & 15;
  const int lhi  = lane >> 4;

  const int tile = blockIdx.x * 8 + wave;
  const int rt = tile >> 5;   // 1024 row tiles
  const int ct = tile & 31;   // 32 col tiles

  const _Float16* xrow = xf16 + (size_t)(rt * 16 + lrow) * (H_ * D_);
  const _Float16* wrow = w2h  + (size_t)(ct * 16 + lrow) * (H_ * D_);

  v8f c = {};
  for (int k0 = 0; k0 < H_ * D_; k0 += 32) {
    v8h lo  = *(const v8h*)(xrow + k0 + lhi * 16);
    v8h hi8 = *(const v8h*)(xrow + k0 + 8 + lhi * 16);
    v16h a;
#pragma unroll
    for (int i = 0; i < 8; ++i) { a[i] = lo[i]; a[8 + i] = hi8[i]; }
    v16h bw = *(const v16h*)(wrow + k0 + lhi * 16);
    c = __builtin_amdgcn_wmma_f32_16x16x32_f16(false, a, false, bw,
                                               (short)0, c, false, false);
  }
#pragma unroll
  for (int j = 0; j < 8; ++j)
    out[(size_t)(rt * 16 + j + lhi * 8) * NH_ + ct * 16 + lrow] = c[j];
}

// ---------------------------------------------------------------------------
extern "C" void kernel_launch(void* const* d_in, const int* in_sizes, int n_in,
                              void* d_out, int out_size, void* d_ws, size_t ws_size,
                              hipStream_t stream) {
  const float* query = (const float*)d_in[0];
  const float* key   = (const float*)d_in[1];
  const float* value = (const float*)d_in[2];
  const int*   mask  = (const int*)d_in[3];
  const float* w_t   = (const float*)d_in[4];
  const float* b_t   = (const float*)d_in[5];
  const float* W0    = (const float*)d_in[6];
  const float* W1    = (const float*)d_in[7];
  const float* W2    = (const float*)d_in[8];

  char* p = (char*)d_ws;
  auto take = [&](size_t nhalf) {
    _Float16* r = (_Float16*)p;
    p += (nhalf * sizeof(_Float16) + 255) & ~(size_t)255;
    return r;
  };
  _Float16* qf16 = take((size_t)B_ * H_ * LQ_ * DK_);  //  4 MB
  _Float16* kf16 = take((size_t)B_ * H_ * LK_ * 32);   // 16 MB (K-padded)
  _Float16* mvT  = take((size_t)B_ * LK_ * D_);        //  8 MB
  _Float16* mT   = take((size_t)B_ * LK_ * D_);        //  8 MB
  _Float16* xf16 = take((size_t)B_ * LQ_ * H_ * D_);   // 32 MB
  _Float16* w2h  = take((size_t)NH_ * H_ * D_);        //  1 MB

  embed_project_kernel<<<B_ * LQ_, E_, 0, stream>>>(query, w_t, b_t, W0, qf16,
                                                    LQ_, 0.25f, 0);
  embed_project_kernel<<<B_ * LK_, E_, 0, stream>>>(key, w_t, b_t, W1, kf16,
                                                    LK_, 1.0f, 1);
  pack_mv_kernel<<<(B_ * LK_ * D_ + 255) / 256, 256, 0, stream>>>(value, mask,
                                                                  mvT, mT);
  cvt_w2_kernel<<<(NH_ * H_ * D_ + 255) / 256, 256, 0, stream>>>(W2, w2h);

  attn_kernel<<<dim3(LQ_ / 64, H_, B_), 256, 0, stream>>>(qf16, kf16, mvT, mT,
                                                          xf16);

  out_gemm_kernel<<<(B_ * LQ_ / 16) * (NH_ / 16) / 8, 256, 0, stream>>>(
      xf16, w2h, (float*)d_out);
}
